// SignLanguageVQVAEModel_44538810860050
// MI455X (gfx1250) — matrix operand order repaired and checked by reading, not compiled, not run on hardware
//
#include <hip/hip_runtime.h>
#include <hip/hip_bf16.h>

// ---------------------------------------------------------------------------
// SignLanguage VQ-VAE forward on gfx1250 (CDNA5, wave32, WMMA bf16,
// async global->LDS staging, fragment-linear recurrent weights)
// ---------------------------------------------------------------------------

typedef __attribute__((ext_vector_type(16))) __bf16 bf16x16;
typedef __attribute__((ext_vector_type(8)))  float  f32x8;
typedef int v4i_t __attribute__((vector_size(16)));   // matches builtin param type

union FragBF { bf16x16 v; unsigned short u[16]; };

#if defined(__has_builtin)
#if __has_builtin(__builtin_amdgcn_global_load_async_to_lds_b128)
#define HAS_ASYNC_LDS 1
#endif
#endif

#ifdef HAS_ASYNC_LDS
__device__ __forceinline__ void async_copy_b128(const void* g, void* l) {
  // memory -> LDS, 16B per lane, tracked by ASYNCcnt (GLOBAL_LOAD_ASYNC_TO_LDS_B128)
  __builtin_amdgcn_global_load_async_to_lds_b128(
      (__attribute__((address_space(1))) v4i_t*)(unsigned long long)g,
      (__attribute__((address_space(3))) v4i_t*)(unsigned)(unsigned long long)l,
      0, 0);
}
__device__ __forceinline__ void wait_async0() {
#if __has_builtin(__builtin_amdgcn_s_wait_asynccnt)
  __builtin_amdgcn_s_wait_asynccnt(0);
#else
  asm volatile("s_wait_asynccnt 0" ::: "memory");
#endif
}
#endif

__device__ __forceinline__ unsigned short f2bf(float f) {
  union { float f; unsigned int u; } x; x.f = f;
  unsigned int r = x.u + 0x7FFFu + ((x.u >> 16) & 1u);   // round-to-nearest-even
  return (unsigned short)(r >> 16);
}

// A-fragment: 16x32 bf16 (MxK). Lane L: half=L>>4, m=L&15.
// Element e: K = 16*half + e
__device__ __forceinline__ bf16x16 load_a_frag(const unsigned short* base,
                                               int row0, int k0, int ld) {
  int lane = threadIdx.x & 31;
  int hf = lane >> 4, m = lane & 15;
  const unsigned short* r = base + (size_t)(row0 + m) * ld + k0;
  FragBF f;
#pragma unroll
  for (int v = 0; v < 8; ++v) { int k = 16*hf + 2*v; f.u[2*v] = r[k]; f.u[2*v+1] = r[k+1]; }
  return f.v;
}

// B-fragment: 32x16 bf16 (KxN), element (k,n) = W[n0+n][k0+k] (W row-major [N,K]).
// Lane L: half=L>>4, n=L&15. Element e: K = 16*half + e
__device__ __forceinline__ bf16x16 load_b_frag(const unsigned short* base,
                                               int n0, int k0, int ld) {
  int lane = threadIdx.x & 31;
  int hf = lane >> 4, n = lane & 15;
  const unsigned short* r = base + (size_t)(n0 + n) * ld + k0;
  FragBF f;
#pragma unroll
  for (int v = 0; v < 8; ++v) { int k = 16*hf + 2*v; f.u[2*v] = r[k]; f.u[2*v+1] = r[k+1]; }
  return f.v;
}

// ---------------------------------------------------------------------------
// Conversion / init kernels
// ---------------------------------------------------------------------------
__global__ void zero_accum_kernel(float* p) {
  if (threadIdx.x == 0 && blockIdx.x == 0) p[0] = 0.f;
}

// x [B=64, T=512, 225] f32 -> A_bf rows n = t*64+b, 256 cols (zero padded)
__global__ void cvt_x_kernel(const float* __restrict__ x, unsigned short* __restrict__ out) {
  int id = blockIdx.x * blockDim.x + threadIdx.x;     // 32768*256 threads
  int n = id >> 8, d = id & 255;
  int t = n >> 6, b = n & 63;
  unsigned short r = 0;
  if (d < 225) r = f2bf(x[((size_t)b * 512 + t) * 225 + d]);
  out[(size_t)n * 256 + d] = r;
}

// generic f32 [rows,cin] -> bf16 [rows,cout] zero-padded
__global__ void cvt_pad_kernel(const float* __restrict__ src, unsigned short* __restrict__ dst,
                               int cin, int cout, int total) {
  int id = blockIdx.x * blockDim.x + threadIdx.x;
  if (id >= total) return;
  int r = id / cout, c = id % cout;
  unsigned short v = 0;
  if (c < cin) v = f2bf(src[(size_t)r * cin + c]);
  dst[id] = v;
}

// Repack Whh f32 [1024,256] into fragment-linear bf16:
// frag(nT 0..63, kT 0..7), lane 0..31, elem e 0..15 stored contiguously:
//   dst[((nT*8+kT)*32 + lane)*16 + e] = bf16(Whh[nT*16 + (lane&15)][kT*32 + 16*(lane>>4) + e])
// so each lane's B-fragment is one contiguous 32B run -> 2x global_load_b128.
__global__ void repack_whh_kernel(const float* __restrict__ src,
                                  unsigned short* __restrict__ dst) {
  int id = blockIdx.x * blockDim.x + threadIdx.x;   // 64*8*32*16 = 262144 threads
  int e = id & 15, lane = (id >> 4) & 31, kT = (id >> 9) & 7, nT = id >> 12;
  int n = nT * 16 + (lane & 15);
  int k = kT * 32 + 16 * (lane >> 4) + e;
  dst[id] = f2bf(src[(size_t)n * 256 + k]);
}

// ---------------------------------------------------------------------------
// WMMA GEMM: C[M,N] = A_bf[M,K] @ W_bf[N,K]^T + bias   (pre = x @ Wih^T + b)
// block tile 128M x 64N, 256 threads (8 waves as 4x2, each wave 2x2 WMMA tiles)
// LDS staging via GLOBAL_LOAD_ASYNC_TO_LDS_B128 when available.
// ---------------------------------------------------------------------------
__global__ __launch_bounds__(256) void gemm_bias_kernel(
    const unsigned short* __restrict__ A, int lda,
    const unsigned short* __restrict__ W, int ldw,
    const float* __restrict__ bias,
    float* __restrict__ C, int ldc, int K) {
  __shared__ unsigned short As[128 * 32];
  __shared__ unsigned short Bs[64 * 32];
  const int tid  = threadIdx.x;
  const int lane = tid & 31, wave = tid >> 5;
  const int wm = wave & 3, wn = wave >> 2;
  const int mBlk = blockIdx.x * 128, nBlk = blockIdx.y * 64;

  f32x8 acc[2][2];
#pragma unroll
  for (int i = 0; i < 2; ++i)
#pragma unroll
    for (int j = 0; j < 2; ++j)
#pragma unroll
      for (int e = 0; e < 8; ++e) acc[i][j][e] = 0.f;

  for (int k0 = 0; k0 < K; k0 += 32) {
    __syncthreads();
#ifdef HAS_ASYNC_LDS
    // A tile: 128 rows x 64B = 512 x 16B chunks; 256 threads x 2
#pragma unroll
    for (int i = 0; i < 2; ++i) {
      int e = tid + i * 256;
      int r = e >> 2, c = e & 3;
      async_copy_b128(A + (size_t)(mBlk + r) * lda + k0 + c * 8, &As[r * 32 + c * 8]);
    }
    // B tile: 64 rows x 64B = 256 x 16B chunks; 1 per thread
    {
      int r = tid >> 2, c = tid & 3;
      async_copy_b128(W + (size_t)(nBlk + r) * ldw + k0 + c * 8, &Bs[r * 32 + c * 8]);
    }
    wait_async0();
#else
    const unsigned int* Ag = (const unsigned int*)A;
    const unsigned int* Wg = (const unsigned int*)W;
    unsigned int* AsU = (unsigned int*)As;
    unsigned int* BsU = (unsigned int*)Bs;
    const int ldaU = lda >> 1, ldwU = ldw >> 1;
#pragma unroll
    for (int i = 0; i < 8; ++i) {            // A tile: 128 rows x 16 dwords
      int e = tid + i * 256; int r = e >> 4, c = e & 15;
      AsU[e] = Ag[(size_t)(mBlk + r) * ldaU + (k0 >> 1) + c];
    }
#pragma unroll
    for (int i = 0; i < 4; ++i) {            // B tile: 64 rows x 16 dwords
      int e = tid + i * 256; int r = e >> 4, c = e & 15;
      BsU[e] = Wg[(size_t)(nBlk + r) * ldwU + (k0 >> 1) + c];
    }
#endif
    __syncthreads();
    bf16x16 a0 = load_a_frag(As, wm * 32,      0, 32);
    bf16x16 a1 = load_a_frag(As, wm * 32 + 16, 0, 32);
    bf16x16 b0 = load_b_frag(Bs, wn * 32,      0, 32);
    bf16x16 b1 = load_b_frag(Bs, wn * 32 + 16, 0, 32);
    acc[0][0] = __builtin_amdgcn_wmma_f32_16x16x32_bf16(false, a0, false, b0, (short)0, acc[0][0], false, false);
    acc[0][1] = __builtin_amdgcn_wmma_f32_16x16x32_bf16(false, a0, false, b1, (short)0, acc[0][1], false, false);
    acc[1][0] = __builtin_amdgcn_wmma_f32_16x16x32_bf16(false, a1, false, b0, (short)0, acc[1][0], false, false);
    acc[1][1] = __builtin_amdgcn_wmma_f32_16x16x32_bf16(false, a1, false, b1, (short)0, acc[1][1], false, false);
  }

  const int hf = lane >> 4, n = lane & 15;
#pragma unroll
  for (int ti = 0; ti < 2; ++ti)
#pragma unroll
    for (int tj = 0; tj < 2; ++tj) {
      int col = nBlk + wn * 32 + tj * 16 + n;
      float bv = bias[col];
#pragma unroll
      for (int r = 0; r < 8; ++r) {
        int row = mBlk + wm * 32 + ti * 16 + 8 * hf + r;
        C[(size_t)row * ldc + col] = acc[ti][tj][r] + bv;
      }
    }
}

// ---------------------------------------------------------------------------
// Persistent LSTM scan. grid=2 (block 0 forward, block 1 backward), 1024 thr.
// Per step: g = h @ Whh^T (WMMA; h bf16 in LDS, Whh fragment-linear bf16 from
// L2 via b128 loads) -> gbuf, then gates + c/h update (c in regs).
// ---------------------------------------------------------------------------
__global__ __launch_bounds__(1024) void lstm_scan_kernel(
    const float* __restrict__ pre_f, const float* __restrict__ pre_b,
    const unsigned short* __restrict__ whhR_f, const unsigned short* __restrict__ whhR_b,
    float* __restrict__ hOut, float* __restrict__ gbufBase) {
  __shared__ unsigned short h_lds[64 * 256];
  const int tid = threadIdx.x;
  const bool rev = (blockIdx.x == 1);
  const float* pre = rev ? pre_b : pre_f;
  const bf16x16* whhFrag = (const bf16x16*)(rev ? whhR_b : whhR_f);
  const int colOff = rev ? 256 : 0;
  float* gbuf = gbufBase + (size_t)blockIdx.x * (64 * 1024);

  for (int i = tid; i < 64 * 256; i += 1024) h_lds[i] = 0;
  float cst[16];
#pragma unroll
  for (int u = 0; u < 16; ++u) cst[u] = 0.f;
  __syncthreads();

  const int lane = tid & 31, wave = tid >> 5;
  const int mBase  = (wave & 3) * 16;   // 4 M tiles cover B=64
  const int nGroup = wave >> 2;         // 8 groups x 8 N tiles cover 4H=1024
  const int hf = lane >> 4, ln = lane & 15;

  for (int s = 0; s < 512; ++s) {
    const int t = rev ? (511 - s) : s;

    f32x8 acc[8];
#pragma unroll
    for (int j = 0; j < 8; ++j)
#pragma unroll
      for (int e = 0; e < 8; ++e) acc[j][e] = 0.f;

#pragma unroll 1
    for (int kk = 0; kk < 8; ++kk) {            // K=256 in steps of 32
      const int k0 = kk * 32;
      bf16x16 a = load_a_frag(h_lds, mBase, k0, 256);
#pragma unroll
      for (int j = 0; j < 8; ++j) {
        // fragment-linear: frag index = (nT*8 + kT)*32 + lane, 32B per lane
        bf16x16 b = whhFrag[(size_t)(((nGroup * 8 + j) * 8 + kk) * 32 + lane)];
        acc[j] = __builtin_amdgcn_wmma_f32_16x16x32_bf16(false, a, false, b, (short)0, acc[j], false, false);
      }
    }

#pragma unroll
    for (int j = 0; j < 8; ++j) {
      int col = nGroup * 128 + j * 16 + ln;
#pragma unroll
      for (int r = 0; r < 8; ++r)
        gbuf[(mBase + 8 * hf + r) * 1024 + col] = acc[j][r];
    }

    // prefetch next timestep's pre slice (256KB) while g settles
    if (s + 1 < 512) {
      const float* pn = pre + (size_t)(rev ? t - 1 : t + 1) * (64 * 1024) + tid * 64;
      __builtin_prefetch(pn, 0, 0);
      __builtin_prefetch(pn + 32, 0, 0);
    }

    __threadfence();
    __syncthreads();

    const float* preT = pre + (size_t)t * (64 * 1024);
#pragma unroll 1
    for (int u = 0; u < 16; ++u) {              // 16384 (b,j) pairs / 1024 thr
      int p = u * 1024 + tid;
      int b = p >> 8, j = p & 255;
      int gi = b * 1024 + j;
      float iv = gbuf[gi]       + preT[gi];
      float fv = gbuf[gi + 256] + preT[gi + 256];
      float gv = gbuf[gi + 512] + preT[gi + 512];
      float ov = gbuf[gi + 768] + preT[gi + 768];
      iv = 1.f / (1.f + __expf(-iv));
      fv = 1.f / (1.f + __expf(-fv));
      gv = tanhf(gv);
      ov = 1.f / (1.f + __expf(-ov));
      float cn = fv * cst[u] + iv * gv;
      cst[u] = cn;
      float hv = ov * tanhf(cn);
      h_lds[b * 256 + j] = f2bf(hv);
      hOut[(size_t)(t * 64 + b) * 512 + colOff + j] = hv;
    }
    __threadfence();
    __syncthreads();
  }
}

// ---------------------------------------------------------------------------
// VQ: per row argmin over 128 codes; write bf16 codebook row (dec input);
// accumulate ||q - z||^2 into accum. One wave per row, 8 rows per block.
// ---------------------------------------------------------------------------
__global__ __launch_bounds__(256) void vq_kernel(
    const float* __restrict__ z, const float* __restrict__ cb,
    unsigned short* __restrict__ qbf, float* __restrict__ accum) {
  const int lane = threadIdx.x & 31, wave = threadIdx.x >> 5;
  const int row = blockIdx.x * 8 + wave;
  const float* zr = z + (size_t)row * 512;
  float zv[16];
#pragma unroll
  for (int i = 0; i < 16; ++i) zv[i] = zr[lane + 32 * i];

  float best = 3.4e38f; int bestIdx = 0;
  for (int code = 0; code < 128; ++code) {
    const float* cr = cb + (size_t)code * 512;
    float s = 0.f;
#pragma unroll
    for (int i = 0; i < 16; ++i) { float d = zv[i] - cr[lane + 32 * i]; s += d * d; }
#pragma unroll
    for (int off = 16; off > 0; off >>= 1) s += __shfl_xor(s, off, 32);
    if (s < best) { best = s; bestIdx = code; }
  }
  const float* cr = cb + (size_t)bestIdx * 512;
  unsigned short* qr = qbf + (size_t)row * 512;
#pragma unroll
  for (int i = 0; i < 16; ++i) qr[lane + 32 * i] = f2bf(cr[lane + 32 * i]);
  if (lane == 0) atomicAdd(accum, best);
}

// ---------------------------------------------------------------------------
// logits = dec_out[:, -1] @ W_cls^T + b_cls ; also write vq_loss scalar
// ---------------------------------------------------------------------------
__global__ void finalize_kernel(const float* __restrict__ dec_out,
                                const float* __restrict__ Wc,
                                const float* __restrict__ bc,
                                const float* __restrict__ accum,
                                float* __restrict__ out) {
  int id = blockIdx.x * blockDim.x + threadIdx.x;
  if (id == 0) out[16000] = accum[0] * 1.25f / (32768.0f * 512.0f);
  if (id >= 16000) return;
  int b = id / 250, cls = id % 250;
  const float* h = dec_out + (size_t)(511 * 64 + b) * 512;
  const float* w = Wc + (size_t)cls * 512;
  float s = bc[cls];
  for (int k = 0; k < 512; ++k) s += h[k] * w[k];
  out[id] = s;
}

// ---------------------------------------------------------------------------
extern "C" void kernel_launch(void* const* d_in, const int* in_sizes, int n_in,
                              void* d_out, int out_size, void* d_ws, size_t ws_size,
                              hipStream_t stream) {
  (void)in_sizes; (void)n_in; (void)out_size; (void)ws_size;

  const float* x        = (const float*)d_in[0];
  const float* eWihF    = (const float*)d_in[1];
  const float* eWhhF    = (const float*)d_in[2];
  const float* ebF      = (const float*)d_in[3];
  const float* eWihB    = (const float*)d_in[4];
  const float* eWhhB    = (const float*)d_in[5];
  const float* ebB      = (const float*)d_in[6];
  const float* codebook = (const float*)d_in[7];
  const float* dWihF    = (const float*)d_in[8];
  const float* dWhhF    = (const float*)d_in[9];
  const float* dbF      = (const float*)d_in[10];
  const float* dWihB    = (const float*)d_in[11];
  const float* dWhhB    = (const float*)d_in[12];
  const float* dbB      = (const float*)d_in[13];
  const float* Wcls     = (const float*)d_in[14];
  const float* bcls     = (const float*)d_in[15];
  float* out = (float*)d_out;

  // ---- workspace layout -------------------------------------------------
  char* ws = (char*)d_ws;
  const size_t A_BF_OFF   = 0;
  const size_t A_BF_SZ    = (size_t)32768 * 512 * 2;       // 32 MB (x pad / quantized)
  const size_t W_SMALL    = (size_t)1024 * 256 * 2;
  const size_t W_BIG      = (size_t)1024 * 512 * 2;
  const size_t WIH_EF_OFF = A_BF_OFF + A_BF_SZ;
  const size_t WIH_EB_OFF = WIH_EF_OFF + W_SMALL;
  const size_t WHH_EF_OFF = WIH_EB_OFF + W_SMALL;          // repacked fragment-linear
  const size_t WHH_EB_OFF = WHH_EF_OFF + W_SMALL;
  const size_t WIH_DF_OFF = WHH_EB_OFF + W_SMALL;
  const size_t WIH_DB_OFF = WIH_DF_OFF + W_BIG;
  const size_t WHH_DF_OFF = WIH_DB_OFF + W_BIG;
  const size_t WHH_DB_OFF = WHH_DF_OFF + W_SMALL;
  const size_t PRE_SZ     = (size_t)512 * 64 * 1024 * 4;   // 128 MB
  const size_t PRE_F_OFF  = WHH_DB_OFF + W_SMALL;
  const size_t PRE_B_OFF  = PRE_F_OFF + PRE_SZ;
  const size_t HOUT_OFF   = PRE_B_OFF + PRE_SZ;            // enc_out then dec_out
  const size_t HOUT_SZ    = (size_t)32768 * 512 * 4;       // 64 MB
  const size_t GBUF_OFF   = HOUT_OFF + HOUT_SZ;            // 2 x 64x1024 f32
  const size_t ACC_OFF    = GBUF_OFF + (size_t)2 * 64 * 1024 * 4;

  unsigned short* A_bf   = (unsigned short*)(ws + A_BF_OFF);
  unsigned short* wihEF  = (unsigned short*)(ws + WIH_EF_OFF);
  unsigned short* wihEB  = (unsigned short*)(ws + WIH_EB_OFF);
  unsigned short* whhEF  = (unsigned short*)(ws + WHH_EF_OFF);
  unsigned short* whhEB  = (unsigned short*)(ws + WHH_EB_OFF);
  unsigned short* wihDF  = (unsigned short*)(ws + WIH_DF_OFF);
  unsigned short* wihDB  = (unsigned short*)(ws + WIH_DB_OFF);
  unsigned short* whhDF  = (unsigned short*)(ws + WHH_DF_OFF);
  unsigned short* whhDB  = (unsigned short*)(ws + WHH_DB_OFF);
  float* preF   = (float*)(ws + PRE_F_OFF);
  float* preB   = (float*)(ws + PRE_B_OFF);
  float* hBuf   = (float*)(ws + HOUT_OFF);
  float* gBuf   = (float*)(ws + GBUF_OFF);
  float* accum  = (float*)(ws + ACC_OFF);

  // ---- pipeline ---------------------------------------------------------
  zero_accum_kernel<<<1, 64, 0, stream>>>(accum);

  cvt_x_kernel<<<32768, 256, 0, stream>>>(x, A_bf);   // x -> [32768,256] bf16

  auto cvt = [&](const float* src, unsigned short* dst, int rows, int cin, int cout) {
    int total = rows * cout;
    cvt_pad_kernel<<<(total + 255) / 256, 256, 0, stream>>>(src, dst, cin, cout, total);
  };
  cvt(eWihF, wihEF, 1024, 225, 256);
  cvt(eWihB, wihEB, 1024, 225, 256);
  cvt(dWihF, wihDF, 1024, 512, 512);
  cvt(dWihB, wihDB, 1024, 512, 512);

  // Whh: f32 -> bf16 fragment-linear layout for the recurrent WMMA
  repack_whh_kernel<<<1024, 256, 0, stream>>>(eWhhF, whhEF);
  repack_whh_kernel<<<1024, 256, 0, stream>>>(eWhhB, whhEB);
  repack_whh_kernel<<<1024, 256, 0, stream>>>(dWhhF, whhDF);
  repack_whh_kernel<<<1024, 256, 0, stream>>>(dWhhB, whhDB);

  dim3 gGemm(32768 / 128, 1024 / 64);
  // encoder input projections: pre = x_pad @ Wih^T + b   (K=256)
  gemm_bias_kernel<<<gGemm, 256, 0, stream>>>(A_bf, 256, wihEF, 256, ebF, preF, 1024, 256);
  gemm_bias_kernel<<<gGemm, 256, 0, stream>>>(A_bf, 256, wihEB, 256, ebB, preB, 1024, 256);

  // encoder bidirectional scan -> hBuf (enc_out [32768, 512])
  lstm_scan_kernel<<<2, 1024, 0, stream>>>(preF, preB, whhEF, whhEB, hBuf, gBuf);

  // vector quantization: enc_out -> A_bf (quantized bf16 [32768,512]) + loss
  vq_kernel<<<32768 / 8, 256, 0, stream>>>(hBuf, codebook, A_bf, accum);

  // decoder input projections (K=512)
  gemm_bias_kernel<<<gGemm, 256, 0, stream>>>(A_bf, 512, wihDF, 512, dbF, preF, 1024, 512);
  gemm_bias_kernel<<<gGemm, 256, 0, stream>>>(A_bf, 512, wihDB, 512, dbB, preB, 1024, 512);

  // decoder bidirectional scan -> hBuf reused as dec_out
  lstm_scan_kernel<<<2, 1024, 0, stream>>>(preF, preB, whhDF, whhDB, hBuf, gBuf);

  // classifier + vq_loss scalar
  finalize_kernel<<<(16000 + 255) / 256, 256, 0, stream>>>(hBuf, Wcls, bcls, accum, out);
}
